// BitLinear_70987219468587
// MI455X (gfx1250) — matrix operand order repaired and checked by reading, not compiled, not run on hardware
//
#include <hip/hip_runtime.h>
#include <stdint.h>

// BitLinear forward for MI455X (gfx1250, wave32, WMMA).
//   out = x @ (sign(W) * mean|W|)^T + bias
// M=8192, N=11008, K=4096.  Compute-bound bf16 WMMA GEMM; packed operands
// (Xb 64MB + Wt 86MB) both live in the 192MB L2 after first touch.
// Tile staging uses CDNA5 async global->LDS DMA (ASYNCcnt) when available.

#define DIN   4096
#define DOUT  11008
#define MTOT  8192          // B*S = 2*4096

typedef __attribute__((ext_vector_type(16))) __bf16 bf16x16;
typedef __attribute__((ext_vector_type(8)))  float  floatx8;
typedef __attribute__((ext_vector_type(4)))  int    i32x4;

struct Q2 { uint4 a, b; };   // 32 bytes == one bf16x16 fragment

#define AS1 __attribute__((address_space(1)))
#define AS3 __attribute__((address_space(3)))

#if __has_builtin(__builtin_amdgcn_global_load_async_to_lds_b128)
#define USE_ASYNC_LDS 1
#else
#define USE_ASYNC_LDS 0
#endif

#if __has_builtin(__builtin_amdgcn_s_wait_asynccnt)
#define WAIT_ASYNC(n) __builtin_amdgcn_s_wait_asynccnt(n)
#else
#define WAIT_ASYNC(n) asm volatile("s_wait_asynccnt %0" ::"i"(n) : "memory")
#endif

__device__ __forceinline__ unsigned short f32_to_bf16_rne(float f) {
  union { float f; unsigned u; } v; v.f = f;
  unsigned r = v.u + 0x7FFFu + ((v.u >> 16) & 1u);
  return (unsigned short)(r >> 16);
}

// ---------------------------------------------------------------- scale ----
__global__ void absmean_reduce(const float* __restrict__ w,
                               float* __restrict__ sumOut, int n) {
  __shared__ float red[256];
  float s = 0.f;
  for (int i = blockIdx.x * blockDim.x + threadIdx.x; i < n;
       i += gridDim.x * blockDim.x)
    s += fabsf(w[i]);
  red[threadIdx.x] = s;
  __syncthreads();
  for (int off = 128; off > 0; off >>= 1) {
    if ((int)threadIdx.x < off) red[threadIdx.x] += red[threadIdx.x + off];
    __syncthreads();
  }
  if (threadIdx.x == 0) atomicAdd(sumOut, red[0]);
}

// ------------------------------------------------------------ x -> bf16 ----
__global__ void convert_x_bf16(const float4* __restrict__ x4,
                               ushort4* __restrict__ xb4, int n4) {
  int i = blockIdx.x * blockDim.x + threadIdx.x;
  if (i < n4) {
    float4 f = x4[i];
    ushort4 o;
    o.x = f32_to_bf16_rne(f.x);
    o.y = f32_to_bf16_rne(f.y);
    o.z = f32_to_bf16_rne(f.z);
    o.w = f32_to_bf16_rne(f.w);
    xb4[i] = o;
  }
}

// -------------------------------------------- W[N,K] -> sign bf16 [K,N] ----
__global__ void sign_transpose(const float* __restrict__ w,
                               unsigned short* __restrict__ wt) {
  __shared__ unsigned short tile[32][33];
  const int k0 = blockIdx.x * 32;
  const int n0 = blockIdx.y * 32;
  const int tx = threadIdx.x;   // 0..31
  const int ty = threadIdx.y;   // 0..7
#pragma unroll
  for (int i = 0; i < 4; ++i) {
    int n = n0 + ty + i * 8;
    float f = w[(size_t)n * DIN + k0 + tx];
    tile[ty + i * 8][tx] =
        (f > 0.f) ? (unsigned short)0x3F80
                  : (f < 0.f ? (unsigned short)0xBF80 : (unsigned short)0);
  }
  __syncthreads();
#pragma unroll
  for (int i = 0; i < 4; ++i) {
    int k = k0 + ty + i * 8;
    wt[(size_t)k * DOUT + n0 + tx] = tile[tx][ty + i * 8];
  }
}

// ------------------------------------------------------------- GEMM --------
#define BM 128
#define BN 128
#define BK 32
#define AS_STRIDE 40    // halfwords: 32 + 8 pad  -> bank-conflict-free frags
#define BS_STRIDE 136   // halfwords: 128 + 8 pad
#if USE_ASYNC_LDS
#define NBUF 3          // nbuf - issue_dist >= 2 -> single barrier per k-step
#else
#define NBUF 2
#endif

__global__ __launch_bounds__(256)
void bitlinear_wmma(const unsigned short* __restrict__ Xb,   // [MTOT][DIN] bf16
                    const unsigned short* __restrict__ Wt,   // [DIN][DOUT] bf16 ±1
                    const float* __restrict__ bias,
                    const float* __restrict__ scaleSum,
                    float* __restrict__ out) {
  __shared__ unsigned short As[NBUF][BM * AS_STRIDE];
  __shared__ unsigned short Bs[NBUF][BK * BS_STRIDE];

  const int tid  = threadIdx.x;
  const int wave = tid >> 5;
  const int lane = tid & 31;
  const int hi   = lane >> 4;    // 0: lanes 0-15, 1: lanes 16-31
  const int ln   = lane & 15;
  const int wm   = wave & 3;     // 4 waves along M
  const int wn   = wave >> 2;    // 2 waves along N

  const int n0 = blockIdx.x * BN;
  const int m0 = blockIdx.y * BM;

  // cooperative tile loaders: 256 threads * 32B = 8KB per tile
  const int ar = tid >> 1, ah = tid & 1;   // A: row 0..127, 32B half of 64B row
  const int br = tid >> 3, bc = tid & 7;   // B: k-row 0..31, 32B chunk of 256B row

  const unsigned short* aSrc = Xb + (size_t)(m0 + ar) * DIN + ah * 16;
  const unsigned short* bSrc = Wt + (size_t)br * DOUT + n0 + bc * 16;
  const int KT = DIN / BK;   // 128 k-steps

  floatx8 acc[2][4];
#pragma unroll
  for (int i = 0; i < 2; ++i)
#pragma unroll
    for (int j = 0; j < 4; ++j) acc[i][j] = {};

#if USE_ASYNC_LDS
  // ---- async global->LDS staging (4 b128 DMA ops / thread / stage) ----
  auto issueStage = [&](int kt, int buf) {
    AS1 i32x4* aS = (AS1 i32x4*)(aSrc + (size_t)kt * BK);
    AS1 i32x4* bS = (AS1 i32x4*)(bSrc + (size_t)kt * BK * DOUT);
    AS3 i32x4* aD = (AS3 i32x4*)&As[buf][ar * AS_STRIDE + ah * 16];
    AS3 i32x4* bD = (AS3 i32x4*)&Bs[buf][br * BS_STRIDE + bc * 16];
    // imm byte-offset applies to both global and LDS addresses (ISA async ch.)
    __builtin_amdgcn_global_load_async_to_lds_b128(aS, aD, 0, 0);
    __builtin_amdgcn_global_load_async_to_lds_b128(aS, aD, 16, 0);
    __builtin_amdgcn_global_load_async_to_lds_b128(bS, bD, 0, 0);
    __builtin_amdgcn_global_load_async_to_lds_b128(bS, bD, 16, 0);
  };

  issueStage(0, 0);
  int cur = 0, nxt = 1;
  for (int kt = 0; kt < KT; ++kt) {
    if (kt + 1 < KT) {
      issueStage(kt + 1, nxt);
      WAIT_ASYNC(4);   // stage kt's 4 per-wave DMA ops have landed (in-order)
    } else {
      WAIT_ASYNC(0);
    }
    __syncthreads();   // stage kt visible to every wave
#else
  // ---- fallback: register-staged double buffer ----
  uint4 a0 = ((const uint4*)aSrc)[0];
  uint4 a1 = ((const uint4*)aSrc)[1];
  uint4 b0 = ((const uint4*)bSrc)[0];
  uint4 b1 = ((const uint4*)bSrc)[1];
  {
    uint4* da = (uint4*)&As[0][ar * AS_STRIDE + ah * 16];
    da[0] = a0; da[1] = a1;
    uint4* db = (uint4*)&Bs[0][br * BS_STRIDE + bc * 16];
    db[0] = b0; db[1] = b1;
  }
  __syncthreads();
  for (int kt = 0; kt < KT; ++kt) {
    const int cur = kt & 1, nxt = cur ^ 1;
    if (kt + 1 < KT) {
      const unsigned short* aS = aSrc + (size_t)(kt + 1) * BK;
      const unsigned short* bS = bSrc + (size_t)(kt + 1) * BK * DOUT;
      a0 = ((const uint4*)aS)[0];
      a1 = ((const uint4*)aS)[1];
      b0 = ((const uint4*)bS)[0];
      b1 = ((const uint4*)bS)[1];
    }
#endif

    // B fragments: lane holds N-col = ln, K rows hi*8+0..7 and 16+hi*8+0..7
    bf16x16 bf[4];
#pragma unroll
    for (int tn = 0; tn < 4; ++tn) {
      const unsigned short* p =
          &Bs[cur][(hi * 8) * BS_STRIDE + wn * 64 + tn * 16 + ln];
      unsigned short t[16];
#pragma unroll
      for (int s = 0; s < 8; ++s) {
        t[s] = p[s * BS_STRIDE];
        t[s + 8] = p[(s + 16) * BS_STRIDE];
      }
      bf[tn] = __builtin_bit_cast(bf16x16, t);
    }
    // A fragments: lane<16 -> k 0-7 & 16-23, lane>=16 -> +8 (ISA 16-bit layout)
    bf16x16 af[2];
#pragma unroll
    for (int tm = 0; tm < 2; ++tm) {
      const unsigned short* p =
          &As[cur][(wm * 32 + tm * 16 + ln) * AS_STRIDE + hi * 8];
      Q2 q;
      q.a = ((const uint4*)p)[0];
      q.b = ((const uint4*)(p + 16))[0];
      af[tm] = __builtin_bit_cast(bf16x16, q);
    }

#pragma unroll
    for (int tm = 0; tm < 2; ++tm)
#pragma unroll
      for (int tn = 0; tn < 4; ++tn)
        acc[tm][tn] = __builtin_amdgcn_wmma_f32_16x16x32_bf16(
            false, af[tm], false, bf[tn], (short)0, acc[tm][tn], false, false);

#if USE_ASYNC_LDS
    cur = (cur == NBUF - 1) ? 0 : cur + 1;
    nxt = (nxt == NBUF - 1) ? 0 : nxt + 1;
  }
#else
    if (kt + 1 < KT) {
      uint4* da = (uint4*)&As[nxt][ar * AS_STRIDE + ah * 16];
      da[0] = a0; da[1] = a1;
      uint4* db = (uint4*)&Bs[nxt][br * BS_STRIDE + bc * 16];
      db[0] = b0; db[1] = b1;
    }
    __syncthreads();
  }
#endif

  // ---- epilogue: scale + bias, C layout: VGPR v -> M = v + hi*8, N = ln ----
  const float scale = scaleSum[0] * (1.0f / ((float)DOUT * (float)DIN));
#pragma unroll
  for (int tm = 0; tm < 2; ++tm) {
    const int rowBase = m0 + wm * 32 + tm * 16 + hi * 8;
#pragma unroll
    for (int tn = 0; tn < 4; ++tn) {
      const int col = n0 + wn * 64 + tn * 16 + ln;
      const float bv = bias[col];
#pragma unroll
      for (int v = 0; v < 8; ++v)
        out[(size_t)(rowBase + v) * DOUT + col] = acc[tm][tn][v] * scale + bv;
    }
  }
}

// ---------------------------------------------------------------------------
extern "C" void kernel_launch(void* const* d_in, const int* in_sizes, int n_in,
                              void* d_out, int out_size, void* d_ws,
                              size_t ws_size, hipStream_t stream) {
  const float* x    = (const float*)d_in[0];   // [2,4096,4096]
  const float* w    = (const float*)d_in[1];   // [11008,4096]
  const float* bias = (const float*)d_in[2];   // [11008]
  float* out = (float*)d_out;                  // [2,4096,11008]

  // workspace: Xb bf16 (64MB) | Wt bf16 (86MB) | scale accumulator
  char* ws = (char*)d_ws;
  unsigned short* Xb = (unsigned short*)ws;
  unsigned short* Wt = (unsigned short*)(ws + (size_t)MTOT * DIN * 2);
  float* scaleSum =
      (float*)(ws + (size_t)MTOT * DIN * 2 + (size_t)DIN * DOUT * 2);

  (void)hipMemsetAsync(scaleSum, 0, sizeof(float), stream);

  absmean_reduce<<<2048, 256, 0, stream>>>(w, scaleSum, DOUT * DIN);

  const int n4 = MTOT * DIN / 4;
  convert_x_bf16<<<(n4 + 255) / 256, 256, 0, stream>>>(
      (const float4*)x, (ushort4*)Xb, n4);

  dim3 tb(32, 8);
  dim3 tg(DIN / 32, DOUT / 32);
  sign_transpose<<<tg, tb, 0, stream>>>(w, Wt);

  dim3 gg(DOUT / BN, MTOT / BM);
  bitlinear_wmma<<<gg, 256, 0, stream>>>(Xb, Wt, bias, scaleSum, out);
}
